// LSAEncoder_24704651887139
// MI455X (gfx1250) — matrix operand order repaired and checked by reading, not compiled, not run on hardware
//
#include <hip/hip_runtime.h>
#include <hip/hip_bf16.h>

// ---------------- problem constants ----------------
#define BB 32          // batch
#define TT 512         // sequence length
#define DD 128         // input dim
#define HH 256         // hidden
#define G4 1024        // 4*H (gates)
#define NTHREADS 512   // 16 wave32 waves
#define BH (BB*HH)     // 8192

// ---------------- workspace layout (bytes) ----------------
// bf16 fused weights: W0 = [Wih0|Whh0] (1024 x 384), W1 = [Wih1|Whh1] (1024 x 512)
#define WS_W0   0u
#define WS_W1   (WS_W0 + 1024u*384u*2u)            // +786432
#define WS_B0   (WS_W1 + 1024u*512u*2u)            // fused bias fp32 (1024)
#define WS_B1   (WS_B0 + 4096u)
#define WS_RAW  (WS_B1 + 4096u)                    // raw_hist  bf16: 2 * T*B*H
#define WS_REF  (WS_RAW + 2u*TT*BB*HH*2u)          // ref_hist  bf16: 2 * T*B*H

// ---------------- LDS layout (bytes), dynamic ----------------
#define LDS_HSTATE 0u                 // fp32 [2][B][H]  = 64 KB (persistent)
#define LDS_CSTATE 65536u             // fp32 [2][B][H]  = 64 KB (persistent)
#define LDS_AIN    131072u            // bf16 [B][512]   = 32 KB (layer input, K<=512)
#define LDS_SCR    163840u            // 128 KB scratch: gates fp32 [B][1024]; aliased attw fp32 [B][T]
#define SMEM_BYTES 294912u            // 288 KB

typedef __attribute__((ext_vector_type(16))) __bf16        bf16x16;
typedef __attribute__((ext_vector_type(8)))  float         f32x8;
typedef __attribute__((ext_vector_type(4)))  unsigned int  u32x4;

__device__ __forceinline__ unsigned short f2bf(float f) {
    unsigned u = __builtin_bit_cast(unsigned, f);
    u += 0x7fffu + ((u >> 16) & 1u);              // round-to-nearest-even
    return (unsigned short)(u >> 16);
}
__device__ __forceinline__ float bf2f(unsigned v) {
    return __builtin_bit_cast(float, v << 16);
}

// 32-byte fragment = two 16B chunks (matches the 16-bit A/B VGPR layouts)
__device__ __forceinline__ bf16x16 load_frag(const unsigned short* p0,
                                             const unsigned short* p1) {
    union { u32x4 v[2]; bf16x16 f; } u;
    u.v[0] = *(const u32x4*)p0;
    u.v[1] = *(const u32x4*)p1;
    return u.f;
}

// ================= weight prep: fuse + fp32->bf16 =================
__global__ void lsa_prep(const float* __restrict__ Wih0, const float* __restrict__ Whh0,
                         const float* __restrict__ bih0, const float* __restrict__ bhh0,
                         const float* __restrict__ Wih1, const float* __restrict__ Whh1,
                         const float* __restrict__ bih1, const float* __restrict__ bhh1,
                         unsigned char* __restrict__ ws) {
    unsigned short* W0 = (unsigned short*)(ws + WS_W0);
    unsigned short* W1 = (unsigned short*)(ws + WS_W1);
    float* b0 = (float*)(ws + WS_B0);
    float* b1 = (float*)(ws + WS_B1);
    const int tid = blockIdx.x * blockDim.x + threadIdx.x;
    const int stride = gridDim.x * blockDim.x;
    for (int e = tid; e < 1024 * 384; e += stride) {
        int n = e / 384, k = e % 384;
        float v = (k < DD) ? Wih0[n * DD + k] : Whh0[n * HH + (k - DD)];
        W0[n * 384 + k] = f2bf(v);
    }
    for (int e = tid; e < 1024 * 512; e += stride) {
        int n = e >> 9, k = e & 511;
        float v = (k < HH) ? Wih1[n * HH + k] : Whh1[n * HH + (k - HH)];
        W1[n * 512 + k] = f2bf(v);
    }
    for (int n = tid; n < G4; n += stride) {
        b0[n] = bih0[n] + bhh0[n];
        b1[n] = bih1[n] + bhh1[n];
    }
}

// ================= gates GEMM: (32 x K)bf16 . (K x 1024)bf16 -> fp32 LDS ======
// A row-major in LDS (stride 512 elems). W row-major (1024 x K) global (L2-resident).
// Fully unrolled k-loop with double-buffered B fragments: loads for tile kt+1 are
// issued before the 8 WMMAs of tile kt so the compiler can use partial loadcnt waits.
template <int KTN>
__device__ __forceinline__ void gemm_gates_t(const unsigned short* __restrict__ A_lds,
                                             const unsigned short* __restrict__ Wg,
                                             float* __restrict__ gates) {
    constexpr int K = KTN * 32;
    const int tid  = threadIdx.x;
    const int wave = tid >> 5, lane = tid & 31;
    const int mt   = wave & 1;        // M tile (0/1)
    const int grp  = wave >> 1;       // 8 N tiles per wave
    const int l16  = lane & 15;
    const int hi   = lane >> 4;       // half-wave select
    f32x8 acc[8];
    const f32x8 zero = {0.f,0.f,0.f,0.f,0.f,0.f,0.f,0.f};
#pragma unroll
    for (int i = 0; i < 8; ++i) acc[i] = zero;

    // A fragment: lane row = mt*16 + l16; elems 0..7 -> kg..kg+7, 8..15 -> kg+16..kg+23
    const unsigned short* abase = A_lds + (mt * 16 + l16) * 512 + (hi << 3);

    // per-output-column base pointers (constant IOFFSETs inside the unrolled loop)
    const unsigned short* pb[8];
#pragma unroll
    for (int i = 0; i < 8; ++i)
        pb[i] = Wg + (size_t)(((((grp << 3) + i) << 4) + l16)) * K + (hi << 3);

    bf16x16 bcur[8];
#pragma unroll
    for (int i = 0; i < 8; ++i) bcur[i] = load_frag(pb[i], pb[i] + 16);

#pragma unroll
    for (int kt = 0; kt < KTN; ++kt) {
        const unsigned short* pa = abase + (kt << 5);
        bf16x16 afrag = load_frag(pa, pa + 16);
        bf16x16 bnxt[8];
        if (kt + 1 < KTN) {
#pragma unroll
            for (int i = 0; i < 8; ++i) {
                const unsigned short* p = pb[i] + ((kt + 1) << 5);
                bnxt[i] = load_frag(p, p + 16);
            }
        }
#pragma unroll
        for (int i = 0; i < 8; ++i)
            acc[i] = __builtin_amdgcn_wmma_f32_16x16x32_bf16(
                         false, afrag, false, bcur[i], (short)0, acc[i], false, false);
        if (kt + 1 < KTN) {
#pragma unroll
            for (int i = 0; i < 8; ++i) bcur[i] = bnxt[i];
        }
    }
    // C layout: VGPR j -> row mt*16 + j + hi*8, col = ntile*16 + l16
    const int crow = mt * 16 + (hi << 3);
#pragma unroll
    for (int i = 0; i < 8; ++i) {
        float* gp = gates + (((grp << 3) + i) << 4) + l16;
#pragma unroll
        for (int j = 0; j < 8; ++j) gp[(crow + j) * G4] = acc[i][j];
    }
}

// ================= persistent encoder kernel =================
extern "C" __global__ void __launch_bounds__(NTHREADS, 1)
lsa_encoder(const float* __restrict__ x, float* __restrict__ out,
            unsigned char* __restrict__ ws) {
    extern __shared__ unsigned char smem[];
    float*          h_state = (float*)(smem + LDS_HSTATE);       // [2][B][H]  h_raw recurrence
    float*          c_state = (float*)(smem + LDS_CSTATE);       // [2][B][H]
    unsigned short* A_in    = (unsigned short*)(smem + LDS_AIN); // [B][512] bf16
    float*          gates   = (float*)(smem + LDS_SCR);          // [B][1024]
    float*          attw    = (float*)(smem + LDS_SCR);          // [B][T] (aliases gates)

    const int tid  = threadIdx.x;
    const int wave = tid >> 5, lane = tid & 31;

    for (int e = tid; e < 2 * BH; e += NTHREADS) { h_state[e] = 0.f; c_state[e] = 0.f; }
    __syncthreads();

    for (int t = 0; t < TT; ++t) {
        for (int l = 0; l < 2; ++l) {
            const float* bias    = (const float*)(ws + (l == 0 ? WS_B0 : WS_B1));
            unsigned short* rawL = (unsigned short*)(ws + WS_RAW) + (size_t)l * TT * BB * HH;
            unsigned short* refL = (unsigned short*)(ws + WS_REF) + (size_t)l * TT * BB * HH;

            // ---- stage layer input A = [in | h_prev] as bf16 in LDS ----
            if (l == 0) {
                for (int e = tid; e < BB * DD; e += NTHREADS) {
                    int b = e >> 7, d = e & 127;
                    const float* xp = x + (size_t)b * TT * DD + (size_t)t * DD + d;
                    A_in[b * 512 + d] = f2bf(*xp);
                    if ((d & 15) == 0 && t + 1 < TT)
                        __builtin_prefetch((const void*)(xp + DD), 0, 3); // next timestep's slice
                }
            }
            const int koff = (l == 0) ? DD : HH;  // (layer1 cols 0..255 already hold h_out of layer0)
            for (int e = tid; e < BH; e += NTHREADS) {
                int b = e >> 8, h = e & 255;
                A_in[b * 512 + koff + h] = f2bf(h_state[l * BH + e]);
            }
            __syncthreads();

            // ---- gates = A . W^T (WMMA, pipelined) ----
            if (l == 0)
                gemm_gates_t<12>(A_in, (const unsigned short*)(ws + WS_W0), gates);
            else
                gemm_gates_t<16>(A_in, (const unsigned short*)(ws + WS_W1), gates);
            __syncthreads();

            // ---- element-wise LSTM cell ----
            for (int e = tid; e < BH; e += NTHREADS) {
                int b = e >> 8, h = e & 255;
                const float* gr = gates + b * G4;
                float gi = gr[h]           + bias[h];
                float gf = gr[HH + h]      + bias[HH + h];
                float gg = gr[2 * HH + h]  + bias[2 * HH + h];
                float go = gr[3 * HH + h]  + bias[3 * HH + h];
                float ig = 1.f / (1.f + __expf(-gi));
                float fg = 1.f / (1.f + __expf(-gf));
                float g  = tanhf(gg);
                float og = 1.f / (1.f + __expf(-go));
                float c  = fg * c_state[l * BH + e] + ig * g;
                c_state[l * BH + e] = c;
                float hr = og * tanhf(c);
                h_state[l * BH + e] = hr;                       // recurrence uses h_raw
                rawL[((size_t)t * BB + b) * HH + h] = f2bf(hr); // attention key history
            }
            __threadfence();
            __syncthreads();

            // ---- causal self-attention over history (wave w owns b = 2w, 2w+1) ----
            float inv[2];
            for (int s = 0; s < 2; ++s) {
                const int b = wave * 2 + s;
                const float* hv = h_state + l * BH + b * HH;
                float lmax = -1e30f;
                if (t > 0) {
                    for (int ti = lane; ti < t; ti += 32) {
                        const unsigned short* rp = rawL + ((size_t)ti * BB + b) * HH;
                        float acc = 0.f;
                        for (int kk = 0; kk < HH; kk += 8) {
                            u32x4 u = *(const u32x4*)(rp + kk);
#pragma unroll
                            for (int j = 0; j < 4; ++j) {
                                acc += bf2f(u[j] & 0xffffu) * hv[kk + 2 * j]
                                     + bf2f(u[j] >> 16)     * hv[kk + 2 * j + 1];
                            }
                        }
                        float sc = acc * 0.0625f;   // 1/sqrt(256)
                        attw[b * TT + ti] = sc;
                        lmax = fmaxf(lmax, sc);
                    }
                }
                for (int m = 16; m >= 1; m >>= 1) lmax = fmaxf(lmax, __shfl_xor(lmax, m, 32));
                float lsum = 0.f;
                if (t > 0) {
                    for (int ti = lane; ti < t; ti += 32) {
                        float ev = __expf(attw[b * TT + ti] - lmax);
                        attw[b * TT + ti] = ev;
                        lsum += ev;
                    }
                }
                for (int m = 16; m >= 1; m >>= 1) lsum += __shfl_xor(lsum, m, 32);
                inv[s] = (t > 0) ? 1.f / lsum : 0.f;
            }
            __syncthreads();   // softmax weights visible before ctx gather

            for (int s = 0; s < 2; ++s) {
                const int b  = wave * 2 + s;
                const int h0 = lane * 8;
                float acc[8] = {0.f,0.f,0.f,0.f,0.f,0.f,0.f,0.f};
                if (t > 0) {
                    for (int ti = 0; ti < t; ++ti) {
                        float wt = attw[b * TT + ti];
                        const unsigned short* rp = refL + ((size_t)ti * BB + b) * HH + h0;
                        u32x4 u = *(const u32x4*)rp;
#pragma unroll
                        for (int j = 0; j < 4; ++j) {
                            acc[2 * j]     += wt * bf2f(u[j] & 0xffffu);
                            acc[2 * j + 1] += wt * bf2f(u[j] >> 16);
                        }
                    }
                }
                const float* hv = h_state + l * BH + b * HH;
                unsigned short pk[8];
#pragma unroll
                for (int j = 0; j < 8; ++j) {
                    float ho = hv[h0 + j] + acc[j] * inv[s];
                    pk[j] = f2bf(ho);
                    if (l == 0) A_in[b * 512 + h0 + j] = pk[j];       // layer1 input
                    else        out[((size_t)b * TT + t) * HH + h0 + j] = ho;
                }
                *(u32x4*)(refL + ((size_t)t * BB + b) * HH + h0) = *(u32x4*)pk;  // value history
            }
            __threadfence();
            __syncthreads();
        }
    }

    // final states: hs, cs appended after encoder_outputs (B*T*H floats)
    const size_t base = (size_t)BB * TT * HH;
    for (int e = tid; e < 2 * BH; e += NTHREADS) {
        out[base + e]          = h_state[e];
        out[base + 2 * BH + e] = c_state[e];
    }
}

extern "C" void kernel_launch(void* const* d_in, const int* in_sizes, int n_in,
                              void* d_out, int out_size, void* d_ws, size_t ws_size,
                              hipStream_t stream) {
    const float* x    = (const float*)d_in[0];
    const float* Wih0 = (const float*)d_in[1];
    const float* Whh0 = (const float*)d_in[2];
    const float* bih0 = (const float*)d_in[3];
    const float* bhh0 = (const float*)d_in[4];
    const float* Wih1 = (const float*)d_in[5];
    const float* Whh1 = (const float*)d_in[6];
    const float* bih1 = (const float*)d_in[7];
    const float* bhh1 = (const float*)d_in[8];

    unsigned char* ws = (unsigned char*)d_ws;

    lsa_prep<<<128, 256, 0, stream>>>(Wih0, Whh0, bih0, bhh0, Wih1, Whh1, bih1, bhh1, ws);

    hipFuncSetAttribute(reinterpret_cast<const void*>(lsa_encoder),
                        hipFuncAttributeMaxDynamicSharedMemorySize, (int)SMEM_BYTES);
    lsa_encoder<<<1, NTHREADS, SMEM_BYTES, stream>>>(x, (float*)d_out, ws);
}
